// mLSTMCell_71073118814390
// MI455X (gfx1250) — matrix operand order repaired and checked by reading, not compiled, not run on hardware
//
#include <hip/hip_runtime.h>
#include <hip/hip_bf16.h>

typedef __attribute__((ext_vector_type(16))) __bf16 v16bf;
typedef __attribute__((ext_vector_type(8)))  float  v8f;
typedef __attribute__((ext_vector_type(4)))  float  f32x4;

#define HDIM 1024
#define BSZ  64

__device__ __forceinline__ f32x4 ld4(const float* p) { return *(const f32x4*)p; }

__device__ __forceinline__ void cvt8(v16bf& d, int base, f32x4 lo, f32x4 hi) {
  d[base + 0] = (__bf16)lo.x; d[base + 1] = (__bf16)lo.y;
  d[base + 2] = (__bf16)lo.z; d[base + 3] = (__bf16)lo.w;
  d[base + 4] = (__bf16)hi.x; d[base + 5] = (__bf16)hi.y;
  d[base + 6] = (__bf16)hi.z; d[base + 7] = (__bf16)hi.w;
}

// Concatenated-activation pointer: columns [0,1024) from A, [1024,2048) from A2.
// kk is always a multiple of 8, and 1024 is a multiple of 8, so an 8-float
// chunk never straddles the segment boundary. Ternary -> v_cndmask (no EXEC
// divergence; WMMA requires EXEC all-ones).
__device__ __forceinline__ const float* cat_ptr(const float* A, const float* A2,
                                                int row, int kk) {
  return (kk < HDIM) ? (A  + ((size_t)row << 10) + kk)
                     : (A2 + ((size_t)row << 10) + (kk - HDIM));
}

// Y[64,1024] = act( concat(A,A2)[64,K] @ W[1024,K]^T + bias )
// Block = 128 threads (4 waves). Each wave owns a 16-wide N tile
// (n0 = blockIdx.x*64 + wave*16) and iterates all four 16-row M tiles,
// reusing the B (weight) fragment across M tiles. grid.x = 1024/64 = 16.
__global__ void __launch_bounds__(128)
gemm64_wmma_bf16(const float* __restrict__ A, const float* __restrict__ A2,
                 int K, const float* __restrict__ W,
                 const float* __restrict__ bias, float* __restrict__ Y, int act)
{
  const int wave = threadIdx.x >> 5;
  const int lane = threadIdx.x & 31;
  const int half = lane >> 4;           // 0: lanes 0-15, 1: lanes 16-31
  const int l15  = lane & 15;
  const int n0   = (blockIdx.x << 6) + (wave << 4);
  const int col  = n0 + l15;            // B column this lane feeds
  const float* wrow = W + (size_t)col * K;

  v8f acc[4] = {v8f{}, v8f{}, v8f{}, v8f{}};

  for (int kb = 0; kb < K; kb += 32) {
    // ---- B fragment (32x16 bf16): lane holds column `col`;
    //      elems 0..7 = K kb+8*half+0..7, elems 8..15 = K kb+16+8*half+0..7.
    v16bf bf_;
    {
      const float* pb = wrow + kb + (half << 3);
      cvt8(bf_, 0, ld4(pb),      ld4(pb + 4));
      cvt8(bf_, 8, ld4(pb + 16), ld4(pb + 20));
    }
    // ---- A fragments per ISA 16-bit A layout:
    //      elems 0..7  = K kb+8*half+0..7, elems 8..15 = K kb+16+8*half+0..7
    const int k0 = kb + (half << 3);
    const int k1 = k0 + 16;
#pragma unroll
    for (int mt = 0; mt < 4; ++mt) {
      const int row = (mt << 4) + l15;
      const float* pa0 = cat_ptr(A, A2, row, k0);
      const float* pa1 = cat_ptr(A, A2, row, k1);
      v16bf af;
      cvt8(af, 0, ld4(pa0), ld4(pa0 + 4));
      cvt8(af, 8, ld4(pa1), ld4(pa1 + 4));
      acc[mt] = __builtin_amdgcn_wmma_f32_16x16x32_bf16(
          false, af, false, bf_, (short)0, acc[mt], false, false);
    }
  }

  const float bn = bias[col];
#pragma unroll
  for (int mt = 0; mt < 4; ++mt) {
#pragma unroll
    for (int vr = 0; vr < 8; ++vr) {
      // D layout: VGPR vr -> M = vr (lanes 0-15) / 8+vr (lanes 16-31)
      const int m = (mt << 4) + vr + (half << 3);
      float val = acc[mt][vr] + bn;
      if (act == 1) val = 1.0f / (1.0f + __expf(-val));
      Y[(size_t)m * HDIM + col] = val;
    }
  }
}

// Fused gated memory update + readout matvec + tanh.
// One wave per (b,m) row of C: C_new[row,:] = f*C + (i*k)*v, and
// r[row] = dot(C_new[row,:], q[b,:]) computed on the fly (saves a full
// 256 MB re-read of C_new). C traffic streamed with non-temporal B128.
__global__ void __launch_bounds__(256)
mlstm_update(const float* __restrict__ C, const float* __restrict__ q,
             const float* __restrict__ k, const float* __restrict__ v,
             const float* __restrict__ fg, const float* __restrict__ ig,
             float* __restrict__ Cnew, float* __restrict__ rt)
{
  const int row  = blockIdx.x * 8 + (threadIdx.x >> 5);   // b*H + m
  const int lane = threadIdx.x & 31;
  const int b    = row >> 10;

  const float fm  = fg[row];
  const float ikm = ig[row] * k[row];
  const float* crow = C    + ((size_t)row << 10);
  float*       orow = Cnew + ((size_t)row << 10);
  const float* vb   = v + (b << 10);
  const float* qb   = q + (b << 10);

  float acc = 0.0f;
#pragma unroll
  for (int n = lane * 4; n < HDIM; n += 128) {
    f32x4 c4 = __builtin_nontemporal_load((const f32x4*)(crow + n));
    f32x4 v4 = *(const f32x4*)(vb + n);   // hot in cache (4 KB per batch row)
    f32x4 q4 = *(const f32x4*)(qb + n);
    f32x4 cn;
    cn.x = fmaf(fm, c4.x, ikm * v4.x);
    cn.y = fmaf(fm, c4.y, ikm * v4.y);
    cn.z = fmaf(fm, c4.z, ikm * v4.z);
    cn.w = fmaf(fm, c4.w, ikm * v4.w);
    __builtin_nontemporal_store(cn, (f32x4*)(orow + n));
    acc = fmaf(cn.x, q4.x, acc);
    acc = fmaf(cn.y, q4.y, acc);
    acc = fmaf(cn.z, q4.z, acc);
    acc = fmaf(cn.w, q4.w, acc);
  }
  // wave32 reduction
#pragma unroll
  for (int off = 16; off; off >>= 1) acc += __shfl_down(acc, off, 32);
  if (lane == 0) rt[row] = tanhf(acc);
}

extern "C" void kernel_launch(void* const* d_in, const int* in_sizes, int n_in,
                              void* d_out, int out_size, void* d_ws, size_t ws_size,
                              hipStream_t stream) {
  (void)in_sizes; (void)n_in; (void)out_size; (void)ws_size;
  const float* x  = (const float*)d_in[0];
  const float* h  = (const float*)d_in[1];
  const float* C  = (const float*)d_in[2];
  const float* Wq = (const float*)d_in[3];  const float* bq = (const float*)d_in[4];
  const float* Wk = (const float*)d_in[5];  const float* bk = (const float*)d_in[6];
  const float* Wv = (const float*)d_in[7];  const float* bv = (const float*)d_in[8];
  const float* Wf = (const float*)d_in[9];  const float* bf = (const float*)d_in[10];
  const float* Wi = (const float*)d_in[11]; const float* bi = (const float*)d_in[12];
  const float* Wo = (const float*)d_in[13]; const float* bo = (const float*)d_in[14];

  float* hnew = (float*)d_out;                 // [B,H]
  float* Cnew = (float*)d_out + BSZ * HDIM;    // [B,H,H]

  float* ws = (float*)d_ws;                    // 6 * 64K floats = 1.5 MB
  float* qb = ws + 0 * BSZ * HDIM;
  float* kb = ws + 1 * BSZ * HDIM;
  float* vb = ws + 2 * BSZ * HDIM;
  float* fb = ws + 3 * BSZ * HDIM;
  float* ib = ws + 4 * BSZ * HDIM;
  float* rt = ws + 5 * BSZ * HDIM;

  // Gate/projection GEMMs (tiny: ~1.1 GFLOP total, weight-bandwidth bound)
  gemm64_wmma_bf16<<<16, 128, 0, stream>>>(x, nullptr, HDIM,       Wq, bq, qb, 0);
  gemm64_wmma_bf16<<<16, 128, 0, stream>>>(x, nullptr, HDIM,       Wk, bk, kb, 0);
  gemm64_wmma_bf16<<<16, 128, 0, stream>>>(x, nullptr, HDIM,       Wv, bv, vb, 0);
  gemm64_wmma_bf16<<<16, 128, 0, stream>>>(x, h,       2 * HDIM,   Wf, bf, fb, 1);
  gemm64_wmma_bf16<<<16, 128, 0, stream>>>(x, h,       2 * HDIM,   Wi, bi, ib, 1);

  // Stream 512 MB of C, fused with readout matvec
  mlstm_update<<<(BSZ * HDIM) / 8, 256, 0, stream>>>(C, qb, kb, vb, fb, ib, Cnew, rt);

  // Output projection
  gemm64_wmma_bf16<<<16, 128, 0, stream>>>(rt, nullptr, HDIM, Wo, bo, hnew, 0);
}
